// C_MoE_59133109732194
// MI455X (gfx1250) — matrix operand order repaired and checked by reading, not compiled, not run on hardware
//
#include <hip/hip_runtime.h>
#include <hip/hip_bf16.h>

#define HW    4096
#define IMG   64
#define BATCH 8
#define ASYNC_LDS 1

typedef __bf16 bf16_t;
typedef bf16_t v16bf __attribute__((ext_vector_type(16)));
typedef bf16_t v8bf  __attribute__((ext_vector_type(8)));
typedef float  v8f   __attribute__((ext_vector_type(8)));

#define CAT16(lo, hi) __builtin_shufflevector(lo, hi, 0,1,2,3,4,5,6,7,8,9,10,11,12,13,14,15)

#if ASYNC_LDS
__device__ __forceinline__ void async_copy_b128(const void* g, const void* l) {
  // async copy global -> LDS, 16 bytes per lane, tracked by ASYNCcnt
  asm volatile("global_load_async_to_lds_b128 %0, %1, off"
               :: "v"((unsigned)(unsigned long long)l),
                  "v"((unsigned long long)g)
               : "memory");
}
__device__ __forceinline__ void wait_async() {
#if __has_builtin(__builtin_amdgcn_s_wait_asynccnt)
  __builtin_amdgcn_s_wait_asynccnt(0);
#else
  asm volatile("s_wait_asynccnt 0" ::: "memory");
#endif
}
#endif

// ---------------------------------------------------------------------------
// f32 -> bf16 elementwise convert (one-time materialization of conv operands)
// ---------------------------------------------------------------------------
__global__ __launch_bounds__(256) void cvt_bf16(
    const float* __restrict__ src, bf16_t* __restrict__ dst, long long n)
{
  const long long stride = (long long)gridDim.x * blockDim.x;
  for (long long i = (long long)blockIdx.x * blockDim.x + threadIdx.x; i < n; i += stride)
    dst[i] = (bf16_t)src[i];
}

// ---------------------------------------------------------------------------
// WMMA implicit-GEMM convolution (1x1 GEMMs and kxk convs via tap loop).
// All operands bf16, f32 accumulate. Output f32 (pre-GN) and/or bf16.
// Y[b, yCoff+o, p] = bias[o] + sum_{tap,c} W[b?][o][c][tap] * X[b, xCoff+c, shift(p)]
// Block tile 128(M) x 256(N); 8 waves, each 64x64 => 16 v_wmma per K-step(32).
// A rows stored K-swizzled [K0-7 | K16-23 | K8-15 | K24-31] so each lane group's
// fragment is one contiguous 32B run (and async b128 chunks scatter directly).
// ---------------------------------------------------------------------------
__global__ __launch_bounds__(256) void conv_wmma(
    const bf16_t* __restrict__ X, const bf16_t* __restrict__ Wt,
    const float* __restrict__ bias, float* __restrict__ Yf, bf16_t* __restrict__ Ybf,
    int Cin, int K, int xCstride, int xCoff, int yCstride, int yCoff,
    long long wBStride)
{
  __shared__ bf16_t As[2][128][40];   // [o][k-swizzled]
  __shared__ bf16_t Bs[2][256][40];   // [p][k], k-contiguous (transposed on load)

  const int tid  = threadIdx.x;
  const int lane = tid & 31;
  const int wv   = tid >> 5;          // 8 waves: 2 (M) x 4 (N)
  const int wmB  = (wv & 1) << 6;     // wave M base: 0 / 64
  const int wnB  = (wv >> 1) << 6;    // wave N base: 0 / 64 / 128 / 192
  const int b     = blockIdx.z;
  const int oBase = blockIdx.y << 7;
  const int pBase = blockIdx.x << 8;
  const int ksq    = K * K;
  const int pad    = K >> 1;
  const int ktiles = Cin >> 5;
  const int total  = ksq * ktiles;
  const bf16_t* Wb = Wt + (long long)b * wBStride;
  const bool gemm1x1 = (ksq == 1);

  v8f acc[4][4];
#pragma unroll
  for (int fm = 0; fm < 4; ++fm)
#pragma unroll
    for (int fn = 0; fn < 4; ++fn)
#pragma unroll
      for (int v = 0; v < 8; ++v) acc[fm][fn][v] = 0.f;

  auto loadTiles = [&](int buf, int it) {
    const int tap = it / ktiles;
    const int k0  = (it - tap * ktiles) << 5;
    {   // A tile: 128 rows x 32 k of bf16 weights, K-swizzled rows
      const int r     = tid >> 1;
      const int kh    = (tid & 1) << 4;       // source K half: 0 / 16
      const int basee = (tid & 1) << 3;       // dest slot base: 0 / 8
      if (gemm1x1) {   // contiguous source: two 16B chunks into swizzled slots
        const bf16_t* src = Wb + (long long)(oBase + r) * Cin + k0 + kh;
#if ASYNC_LDS
        async_copy_b128(src,     &As[buf][r][basee]);
        async_copy_b128(src + 8, &As[buf][r][basee + 16]);
#else
        const uint4* s4 = (const uint4*)src;
        *(uint4*)&As[buf][r][basee]      = s4[0];
        *(uint4*)&As[buf][r][basee + 16] = s4[1];
#endif
      } else {          // stride-ksq gather into swizzled slots
        const bf16_t* wp = Wb + ((long long)(oBase + r) * Cin + (k0 + kh)) * ksq + tap;
        bf16_t* dst = &As[buf][r][0];
#pragma unroll
        for (int i = 0; i < 16; ++i) {
          const int pos = basee + (i < 8 ? i : i + 8);
          dst[pos] = wp[i * ksq];
        }
      }
    }
    {   // B tile: 32 k x 256 p, stored transposed [p][k], spatially shifted
      const int dy  = tap / K - pad;
      const int dx  = tap - (tap / K) * K - pad;
      const int c2  = (tid >> 4) << 1;      // channel pair base 0..30
      const int pl0 = (tid & 15) << 4;      // 16 pixels per thread
      const bf16_t* x0 = X + ((long long)b * xCstride + xCoff + k0 + c2) * HW;
      const bf16_t* x1 = x0 + HW;
      __builtin_prefetch(x0 + 2 * HW, 0, 0);   // stream next channel pair
#pragma unroll
      for (int i = 0; i < 16; ++i) {
        const int p  = pBase + pl0 + i;
        const int yy = (p >> 6) + dy;
        const int xx = (p & 63) + dx;
        const bool ok = ((unsigned)yy < 64u) & ((unsigned)xx < 64u);
        const int off = yy * IMG + xx;
        const bf16_t z = (bf16_t)0.f;
        const bf16_t v0 = ok ? x0[off] : z;
        const bf16_t v1 = ok ? x1[off] : z;
        bf16_t* d = &Bs[buf][pl0 + i][c2];
        d[0] = v0;
        d[1] = v1;
      }
    }
  };

  loadTiles(0, 0);
#if ASYNC_LDS
  if (gemm1x1) wait_async();
#endif
  __syncthreads();

  const int lrow = lane & 15;
  const int kg   = lane >> 4;

  for (int it = 0; it < total; ++it) {
    if (it + 1 < total) loadTiles((it + 1) & 1, it + 1);
    const int buf = it & 1;

    // A frags: contiguous 32B per lane group thanks to the K-swizzle
    v16bf afr[4];
#pragma unroll
    for (int fm = 0; fm < 4; ++fm) {
      const v8bf lo = *(const v8bf*)&As[buf][wmB + (fm << 4) + lrow][kg << 4];
      const v8bf hi = *(const v8bf*)&As[buf][wmB + (fm << 4) + lrow][(kg << 4) + 8];
      afr[fm] = CAT16(lo, hi);
    }
    // B frags loaded one at a time; amortized over 4 M-frags to bound registers
#pragma unroll
    for (int fn = 0; fn < 4; ++fn) {
      const v8bf lo = *(const v8bf*)&Bs[buf][wnB + (fn << 4) + lrow][kg << 3];
      const v8bf hi = *(const v8bf*)&Bs[buf][wnB + (fn << 4) + lrow][(kg << 3) + 16];
      const v16bf bb = CAT16(lo, hi);
#pragma unroll
      for (int fm = 0; fm < 4; ++fm)
        acc[fm][fn] = __builtin_amdgcn_wmma_f32_16x16x32_bf16(
            false, afr[fm], false, bb, (short)0, acc[fm][fn], false, false);
    }
#if ASYNC_LDS
    if (gemm1x1) wait_async();
#endif
    __syncthreads();
  }

  // Epilogue: C layout -> o = base + (lane>>4)*8 + v, p = base + (lane&15)
  const int og = (lane >> 4) << 3;
  const int nl = lane & 15;
#pragma unroll
  for (int fm = 0; fm < 4; ++fm) {
#pragma unroll
    for (int v = 0; v < 8; ++v) {
      const int o = oBase + wmB + (fm << 4) + og + v;
      const float bv = bias ? bias[o] : 0.f;
      const long long rowoff = ((long long)b * yCstride + yCoff + o) * HW + pBase + wnB + nl;
      if (Yf) {
        float* yr = Yf + rowoff;
#pragma unroll
        for (int fn = 0; fn < 4; ++fn) yr[fn << 4] = acc[fm][fn][v] + bv;
      } else {
        bf16_t* yr = Ybf + rowoff;
#pragma unroll
        for (int fn = 0; fn < 4; ++fn) yr[fn << 4] = (bf16_t)(acc[fm][fn][v] + bv);
      }
    }
  }
}

// ---------------------------------------------------------------------------
// GroupNorm statistics: one block per (batch, group); 32 groups always.
// ---------------------------------------------------------------------------
__global__ __launch_bounds__(256) void gn_stats(
    const float* __restrict__ Yp, float* __restrict__ mu, float* __restrict__ rs, int C)
{
  const int bg  = blockIdx.x;              // = b*32 + g ; channel base = bg*cpg
  const int cpg = C >> 5;
  const long long base = (long long)bg * cpg * HW;
  const int n = cpg * HW;
  float s = 0.f, s2 = 0.f;
  for (int i = threadIdx.x; i < n; i += 256) {
    const float v = Yp[base + i];
    s += v; s2 += v * v;
  }
  __shared__ float sh1[256], sh2[256];
  sh1[threadIdx.x] = s; sh2[threadIdx.x] = s2;
  __syncthreads();
  for (int off = 128; off > 0; off >>= 1) {
    if (threadIdx.x < off) {
      sh1[threadIdx.x] += sh1[threadIdx.x + off];
      sh2[threadIdx.x] += sh2[threadIdx.x + off];
    }
    __syncthreads();
  }
  if (threadIdx.x == 0) {
    const float inv = 1.f / (float)n;
    const float m   = sh1[0] * inv;
    const float var = sh2[0] * inv - m * m;
    mu[bg] = m;
    rs[bg] = rsqrtf(var + 1e-5f);
  }
}

// ---------------------------------------------------------------------------
// GroupNorm apply + affine + ReLU; bf16 and/or f32 destinations (nullable),
// scatter into channel-offset slice of destination.
// ---------------------------------------------------------------------------
__global__ __launch_bounds__(256) void gn_apply_relu(
    const float* __restrict__ Yp, const float* __restrict__ mu, const float* __restrict__ rs,
    const float* __restrict__ gam, const float* __restrict__ bet,
    bf16_t* __restrict__ OutB, float* __restrict__ OutF,
    int C, int oCs, int oCo, long long total)
{
  const int cpg = C >> 5;
  const long long stride = (long long)gridDim.x * blockDim.x;
  for (long long i = (long long)blockIdx.x * blockDim.x + threadIdx.x; i < total; i += stride) {
    const int p = (int)(i & (HW - 1));
    const long long bc = i >> 12;
    const int c  = (int)(bc % C);
    const int bb = (int)(bc / C);
    const int bg = (int)(bc / cpg);
    float v = (Yp[i] - mu[bg]) * rs[bg] * gam[c] + bet[c];
    v = v > 0.f ? v : 0.f;
    const long long oi = ((long long)bb * oCs + oCo + c) * HW + p;
    if (OutB) OutB[oi] = (bf16_t)v;
    if (OutF) OutF[oi] = v;
  }
}

// ---------------------------------------------------------------------------
// gml = mean over H,W (f32 source); one block per (b,c)
// ---------------------------------------------------------------------------
__global__ __launch_bounds__(256) void mean_hw(const float* __restrict__ X, float* __restrict__ out)
{
  const long long base = (long long)blockIdx.x * HW;
  float s = 0.f;
  for (int i = threadIdx.x; i < HW; i += 256) s += X[base + i];
  __shared__ float sh[256];
  sh[threadIdx.x] = s; __syncthreads();
  for (int off = 128; off > 0; off >>= 1) {
    if (threadIdx.x < off) sh[threadIdx.x] += sh[threadIdx.x + off];
    __syncthreads();
  }
  if (threadIdx.x == 0) out[blockIdx.x] = sh[0] * (1.f / (float)HW);
}

// ---------------------------------------------------------------------------
// Router: logits -> softmax -> top-2 -> normalized scores; aux loss (lb + z);
// coef[b][n][c] = sum_j scoreN[b,j] * softmax_n(experts[idx[b,j],n,c])
// ---------------------------------------------------------------------------
__global__ __launch_bounds__(256) void gate_coef(
    const float* __restrict__ gml, const float* __restrict__ gw, const float* __restrict__ gb,
    const float* __restrict__ experts, float* __restrict__ coef, float* __restrict__ lb_out)
{
  __shared__ float probs[8][32];
  __shared__ float snorm[8][2];
  __shared__ int   sidx[8][2];
  __shared__ float zterm[8];
  const int tid = threadIdx.x;
  const int b = tid >> 5, r = tid & 31;

  {
    float a = gb[r];
    const float* gv = gml + b * 256;
    const float* wv = gw + r * 256;
    for (int c = 0; c < 256; ++c) a += gv[c] * wv[c];
    probs[b][r] = a;                           // logits, in-place
  }
  __syncthreads();
  if (tid < 8) {
    const int bb = tid;
    float mx = -1e30f;
    for (int j = 0; j < 32; ++j) mx = fmaxf(mx, probs[bb][j]);
    float e[32], se = 0.f;
    for (int j = 0; j < 32; ++j) { e[j] = expf(probs[bb][j] - mx); se += e[j]; }
    const float lse = mx + logf(se);
    zterm[bb] = lse * lse;
    const float inv = 1.f / se;
    for (int j = 0; j < 32; ++j) probs[bb][j] = e[j] * inv;
    int i0 = 0; float v0 = probs[bb][0];
    for (int j = 1; j < 32; ++j) if (probs[bb][j] > v0) { v0 = probs[bb][j]; i0 = j; }
    int i1 = (i0 == 0) ? 1 : 0; float v1 = probs[bb][i1];
    for (int j = 0; j < 32; ++j)
      if (j != i0 && probs[bb][j] > v1) { v1 = probs[bb][j]; i1 = j; }
    const float ssum = v0 + v1 + 1e-8f;
    snorm[bb][0] = v0 / ssum; snorm[bb][1] = v1 / ssum;
    sidx[bb][0] = i0; sidx[bb][1] = i1;
  }
  __syncthreads();
  if (tid == 0) {
    float imp[32], ms = 0.f;
    for (int j = 0; j < 32; ++j) {
      float s = 0.f;
      for (int bb = 0; bb < 8; ++bb) s += probs[bb][j];
      imp[j] = s * 0.125f; ms += imp[j];
    }
    ms *= (1.f / 32.f);
    float var = 0.f;
    for (int j = 0; j < 32; ++j) { const float d = imp[j] - ms; var += d * d; }
    var *= (1.f / 32.f);
    float z = 0.f;
    for (int bb = 0; bb < 8; ++bb) z += zterm[bb];
    z = (z * 0.125f) * 1e-4f;
    lb_out[0] = var / (ms * ms + 1e-8f) + z;
  }
  for (int pc = tid; pc < 8 * 256; pc += 256) {
    const int bb = pc >> 8, c = pc & 255;
    float o[8];
#pragma unroll
    for (int n = 0; n < 8; ++n) o[n] = 0.f;
#pragma unroll
    for (int j = 0; j < 2; ++j) {
      const int e = sidx[bb][j];
      const float sw = snorm[bb][j];
      float v[8]; float mx = -1e30f;
#pragma unroll
      for (int n = 0; n < 8; ++n) { v[n] = experts[(e * 8 + n) * 256 + c]; mx = fmaxf(mx, v[n]); }
      float se = 0.f;
#pragma unroll
      for (int n = 0; n < 8; ++n) { v[n] = expf(v[n] - mx); se += v[n]; }
      const float iv = sw / se;
#pragma unroll
      for (int n = 0; n < 8; ++n) o[n] += v[n] * iv;
    }
#pragma unroll
    for (int n = 0; n < 8; ++n) coef[(bb * 8 + n) * 256 + c] = o[n];
  }
}

// ---------------------------------------------------------------------------
// Wr[b][co][ci][t] = sum_n coef[b][n][co] * base[n][co][ci][t]   (bf16 out)
// ---------------------------------------------------------------------------
__global__ __launch_bounds__(256) void build_wr(
    const float* __restrict__ coef, const float* __restrict__ base,
    bf16_t* __restrict__ wr, int ksq, long long total)
{
  const long long stride = (long long)gridDim.x * blockDim.x;
  for (long long i = (long long)blockIdx.x * blockDim.x + threadIdx.x; i < total; i += stride) {
    const int t = (int)(i % ksq);
    long long rest = i / ksq;
    const int ci = (int)(rest & 255); rest >>= 8;
    const int co = (int)(rest & 255);
    const int bb = (int)(rest >> 8);
    const float* cf = coef + (long long)bb * 8 * 256 + co;
    float a = 0.f;
#pragma unroll
    for (int n = 0; n < 8; ++n)
      a += cf[n * 256] * base[((long long)(n * 256 + co) * 256 + ci) * ksq + t];
    wr[i] = (bf16_t)a;
  }
}

__global__ void finalize_lb(const float* __restrict__ lbs, float* __restrict__ out)
{
  if (threadIdx.x == 0) out[0] = lbs[0] + lbs[1] + lbs[2];
}

// ---------------------------------------------------------------------------
extern "C" void kernel_launch(void* const* d_in, const int* in_sizes, int n_in,
                              void* d_out, int out_size, void* d_ws, size_t ws_size,
                              hipStream_t stream)
{
  (void)n_in; (void)out_size; (void)ws_size;
  auto F = [&](int i) { return (const float*)d_in[i]; };

  const float *share, *related;
  const float *cs_w,*cs_b,*cs_g,*cs_bt, *cp_w,*cp_b,*cp_g,*cp_bt;
  const float *c1_w,*c1_b,*c1_g,*c1_bt, *c2_w,*c2_b,*c2_g,*c2_bt;
  const float *o_w,*o_b,*o_g,*o_bt;
  const float *m_soft[3],*m_gw[3],*m_gb[3],*m_ex[3];
  const float *m_prew[3],*m_preb[3],*m_preg[3],*m_prebt[3];
  const float *m_pow[3],*m_pob[3],*m_pog[3],*m_pobt[3];
  const float *bases[3];

  if (in_sizes[0] == BATCH * 512 * HW) {
    // dict insertion order: share, related, params{...}
    share = F(0); related = F(1);
    cs_w=F(2);  cs_b=F(3);  cs_g=F(4);  cs_bt=F(5);
    cp_w=F(6);  cp_b=F(7);  cp_g=F(8);  cp_bt=F(9);
    c1_w=F(10); c1_b=F(11); c1_g=F(12); c1_bt=F(13);
    c2_w=F(14); c2_b=F(15); c2_g=F(16); c2_bt=F(17);
    o_w=F(18);  o_b=F(19);  o_g=F(20);  o_bt=F(21);
    for (int m = 0; m < 3; ++m) {
      const int q = 22 + m * 12;
      m_soft[m]=F(q);   m_gw[m]=F(q+1);   m_gb[m]=F(q+2);   m_ex[m]=F(q+3);
      m_prew[m]=F(q+4); m_preb[m]=F(q+5); m_preg[m]=F(q+6); m_prebt[m]=F(q+7);
      m_pow[m]=F(q+8);  m_pob[m]=F(q+9);  m_pog[m]=F(q+10); m_pobt[m]=F(q+11);
    }
    bases[0]=F(58); bases[1]=F(59); bases[2]=F(60);
  } else {
    // jax tree_flatten (sorted-key) order: params < related < share
    bases[0]=F(0); bases[1]=F(1); bases[2]=F(2);
    c1_b=F(3);  c1_bt=F(4);  c1_g=F(5);  c1_w=F(6);
    c2_b=F(7);  c2_bt=F(8);  c2_g=F(9);  c2_w=F(10);
    cp_b=F(11); cp_bt=F(12); cp_g=F(13); cp_w=F(14);
    cs_b=F(15); cs_bt=F(16); cs_g=F(17); cs_w=F(18);
    for (int m = 0; m < 3; ++m) {
      const int q = 19 + m * 12;
      m_ex[m]=F(q);     m_gb[m]=F(q+1);   m_gw[m]=F(q+2);
      m_pob[m]=F(q+3);  m_pobt[m]=F(q+4); m_pog[m]=F(q+5);  m_pow[m]=F(q+6);
      m_preb[m]=F(q+7); m_prebt[m]=F(q+8);m_preg[m]=F(q+9); m_prew[m]=F(q+10);
      m_soft[m]=F(q+11);
    }
    o_b=F(55); o_bt=F(56); o_g=F(57); o_w=F(58);
    related=F(59); share=F(60);
  }

  // ---------------- workspace carve (bytes) ----------------
  char* base = (char*)d_ws;
  float*  TMP  = (float*) (base);                  // 16,777,216 f32 (pre-norm scratch)
  bf16_t* KQ   = (bf16_t*)(base +  67108864ll);    // 16,777,216 bf16 (key|query, 512ch)
  bf16_t* XM   = (bf16_t*)(base + 100663296ll);    //  8,388,608 bf16 (cond / pre out)
  bf16_t* CVO  = (bf16_t*)(base + 117440512ll);    // 16,777,216 bf16 (fixed|routed, 512ch)
  bf16_t* OUTS = (bf16_t*)(base + 150994944ll);    // 25,165,824 bf16 (3x256 moe outs)
  bf16_t* SHB  = (bf16_t*)(base + 201326592ll);    // 16,777,216 bf16 share
  bf16_t* RLB  = (bf16_t*)(base + 234881024ll);    // 16,777,216 bf16 related
  bf16_t* WGT  = (bf16_t*)(base + 268435456ll);    //  4,784,128 bf16 weight arena
  bf16_t* WR   = (bf16_t*)(base + 278003712ll);    // 13,107,200 bf16 routed kernels
  float*  GML  = (float*) (base + 304218112ll);    // 2048
  float*  COEF = GML + 2048;                       // 16384
  float*  MU   = COEF + 16384;                     // 256
  float*  RS   = MU + 256;                         // 256
  float*  LBS  = RS + 256;                         // 3

  const dim3 blk(256);
  auto cvt = [&](const float* s, bf16_t* d, long long n) {
    cvt_bf16<<<dim3(2048), blk, 0, stream>>>(s, d, n);
  };

  // one-time bf16 materialization: inputs + all conv weights
  cvt(share,   SHB, 16777216ll);
  cvt(related, RLB, 16777216ll);
  const float* wsrc[14] = { cs_w, cp_w, c1_w, c2_w, o_w,
                            m_soft[0], m_soft[1], m_soft[2],
                            m_prew[0], m_prew[1], m_prew[2],
                            m_pow[0],  m_pow[1],  m_pow[2] };
  const long long wcnt[14] = { 131072, 131072, 1179648, 65536, 393216,
                               65536, 589824, 1638400,
                               65536, 65536, 65536,
                               131072, 131072, 131072 };
  long long woff[14]; { long long o = 0;
    for (int i = 0; i < 14; ++i) { woff[i] = o; cvt(wsrc[i], WGT + o, wcnt[i]); o += wcnt[i]; } }
  bf16_t* W_cs = WGT + woff[0];  bf16_t* W_cp = WGT + woff[1];
  bf16_t* W_c1 = WGT + woff[2];  bf16_t* W_c2 = WGT + woff[3];
  bf16_t* W_o  = WGT + woff[4];
  bf16_t* W_soft[3] = { WGT + woff[5], WGT + woff[6], WGT + woff[7] };
  bf16_t* W_pre[3]  = { WGT + woff[8], WGT + woff[9], WGT + woff[10] };
  bf16_t* W_post[3] = { WGT + woff[11], WGT + woff[12], WGT + woff[13] };

  auto conv = [&](const bf16_t* Xp, const bf16_t* Wp, const float* bp,
                  float* Yf, bf16_t* Ybf,
                  int Cin, int Cout, int K, int xCs, int xCo, int yCs, int yCo,
                  long long wBS) {
    dim3 grid(HW / 256, Cout / 128, BATCH);
    conv_wmma<<<grid, blk, 0, stream>>>(Xp, Wp, bp, Yf, Ybf, Cin, K, xCs, xCo, yCs, yCo, wBS);
  };
  auto cna = [&](const bf16_t* Xp, const bf16_t* Wp, const float* bp, const float* gp,
                 const float* btp, bf16_t* OutB, float* OutF, int Cin, int Cout, int K,
                 int xCs, int xCo, int oCs, int oCo) {
    conv(Xp, Wp, bp, TMP, nullptr, Cin, Cout, K, xCs, xCo, Cout, 0, 0);
    gn_stats<<<dim3(BATCH * 32), blk, 0, stream>>>(TMP, MU, RS, Cout);
    const long long tot = (long long)BATCH * Cout * HW;
    gn_apply_relu<<<dim3(4096), blk, 0, stream>>>(TMP, MU, RS, gp, btp, OutB, OutF,
                                                  Cout, oCs, oCo, tot);
  };

  // key / query -> KQ[0:256) / KQ[256:512)
  cna(SHB, W_cs, cs_b, cs_g, cs_bt, KQ, nullptr, 512, 256, 1, 512, 0, 512, 0);
  cna(RLB, W_cp, cp_b, cp_g, cp_bt, KQ, nullptr, 512, 256, 1, 512, 0, 512, 256);
  // cond = cna3x3(concat) ; cond2 (bf16 + in-place f32 for the router)
  cna(KQ, W_c1, c1_b, c1_g, c1_bt, XM, nullptr, 512, 256, 3, 512, 0, 256, 0);
  cna(XM, W_c2, c2_b, c2_g, c2_bt, XM, TMP,     256, 256, 1, 256, 0, 256, 0);
  mean_hw<<<dim3(BATCH * 256), blk, 0, stream>>>(TMP, GML);

  const int Ks[3] = {1, 3, 5};
  for (int m = 0; m < 3; ++m) {
    const int K = Ks[m], ksq = K * K;
    // pre(key) -> XM
    cna(KQ, W_pre[m], m_preb[m], m_preg[m], m_prebt[m], XM, nullptr, 256, 256, 1, 512, 0, 256, 0);
    gate_coef<<<dim3(1), blk, 0, stream>>>(GML, m_gw[m], m_gb[m], m_ex[m], COEF, LBS + m);
    const long long wrTot = 8ll * 256 * 256 * ksq;
    build_wr<<<dim3(8192), blk, 0, stream>>>(COEF, bases[m], WR, ksq, wrTot);
    // fixed expert (shared weights) -> CVO[0:256)
    conv(XM, W_soft[m], nullptr, nullptr, CVO, 256, 256, K, 256, 0, 512, 0, 0);
    // combined score-weighted routed expert (per-batch weights) -> CVO[256:512)
    conv(XM, WR, nullptr, nullptr, CVO, 256, 256, K, 256, 0, 512, 256, 256ll * 256 * ksq);
    // post cna -> OUTS channel slice m*256
    cna(CVO, W_post[m], m_pob[m], m_pog[m], m_pobt[m], OUTS, nullptr, 512, 256, 1, 512, 0, 768, m * 256);
  }
  // final 768 -> 512 cna straight into d_out (f32)
  cna(OUTS, W_o, o_b, o_g, o_bt, nullptr, (float*)d_out, 768, 512, 1, 768, 0, 512, 0);
  finalize_lb<<<dim3(1), dim3(1), 0, stream>>>(LBS, (float*)d_out + 16777216ll);
}
